// GCN_22247930593397
// MI455X (gfx1250) — matrix operand order repaired and checked by reading, not compiled, not run on hardware
//
#include <hip/hip_runtime.h>

typedef __bf16 bf16;
typedef __attribute__((ext_vector_type(16))) __bf16 v16bf;
typedef __attribute__((ext_vector_type(8)))  float  v8f;

#define Nn 50000
#define En 800000
#define FIN 128
#define Hn 256
#define Gn 512

// encoding of -inf under the monotonic map below: ~0xFF800000 = 0x007FFFFF
#define ENC_NEG_INF 0x007FFFFFu

__device__ inline v16bf make_v16(uint4 lo, uint4 hi) {
  union { uint4 q[2]; v16bf v; } u;
  u.q[0] = lo; u.q[1] = hi;
  return u.v;
}

// Monotonic float -> uint encoding for atomicMax-based segment_max
__device__ inline unsigned enc_f32(float f) {
  unsigned u = __float_as_uint(f);
  return (u & 0x80000000u) ? ~u : (u | 0x80000000u);
}
__device__ inline float dec_f32(unsigned e) {
  unsigned u = (e & 0x80000000u) ? (e & 0x7FFFFFFFu) : ~e;
  return __uint_as_float(u);
}

// ---------------- prep kernels ----------------
__global__ void fill_f32_k(float* __restrict__ p, float v, long n) {
  long i = blockIdx.x * (long)blockDim.x + threadIdx.x;
  long st = (long)gridDim.x * blockDim.x;
  for (; i < n; i += st) p[i] = v;
}

__global__ void fill_u32_k(unsigned* __restrict__ p, unsigned v, long n) {
  long i = blockIdx.x * (long)blockDim.x + threadIdx.x;
  long st = (long)gridDim.x * blockDim.x;
  for (; i < n; i += st) p[i] = v;
}

__global__ void f32_to_bf16_k(const float* __restrict__ in, bf16* __restrict__ out, long n) {
  long i = blockIdx.x * (long)blockDim.x + threadIdx.x;
  long st = (long)gridDim.x * blockDim.x;
  for (; i < n; i += st) out[i] = (bf16)in[i];
}

// W[K][Nc] (row-major f32) -> Wt[Nc][K] (row-major bf16)
__global__ void transpose_bf16_k(const float* __restrict__ W, bf16* __restrict__ Wt,
                                 int K, int Nc) {
  int idx = blockIdx.x * blockDim.x + threadIdx.x;
  if (idx < K * Nc) {
    int n = idx / K, k = idx % K;
    Wt[idx] = (bf16)W[k * Nc + n];
  }
}

__global__ void deg_edges_k(float* __restrict__ deg, const int* __restrict__ dst, int E) {
  int e = blockIdx.x * blockDim.x + threadIdx.x;
  if (e < E) atomicAdd(&deg[dst[e]], 1.0f);
}

__global__ void rsqrt_inplace_k(float* __restrict__ d, int n) {
  int i = blockIdx.x * blockDim.x + threadIdx.x;
  if (i < n) {
    float v = d[i];
    d[i] = (v > 0.0f) ? rsqrtf(v) : 0.0f;
  }
}

// ---------------- WMMA GEMM: C[M x 256] = A[M x K](bf16) @ Bt[256 x K]^T ----------------
// grid = (M/16, 2), block = 256 (8 waves); wave w covers n-tile blockIdx.y*8 + w.
// Operand layouts per CDNA5 ISA 7.12.2 (wave32):
//  A (16x32 bf16): lane L holds row L&15; K chunks (L>>4)*8+{0..7} and 16+(L>>4)*8+{0..7}
//  B (32x16 bf16): lane L holds col L&15; K chunks (L>>4)*8+{0..7} and 16+(L>>4)*8+{0..7}
//  D (16x16 f32) : VGPR v, lanes 0-15 -> M=v, lanes 16-31 -> M=8+v; N=L&15
__global__ void gemm_bf16_wmma_k(const bf16* __restrict__ A, const bf16* __restrict__ Bt,
                                 float* __restrict__ C, int K) {
  const int lane = threadIdx.x & 31;
  const int wave = threadIdx.x >> 5;
  const int m0 = blockIdx.x * 16;
  const int n0 = (blockIdx.y * 8 + wave) * 16;
  const int half = lane >> 4;              // 0 or 1
  const int lid  = lane & 15;
  const bf16* arow = A  + (size_t)(m0 + lid) * K;
  const bf16* brow = Bt + (size_t)(n0 + lid) * K;
  const int akoff = half * 8;
  const int bkoff = half * 8;

  v8f acc = {0.f, 0.f, 0.f, 0.f, 0.f, 0.f, 0.f, 0.f};
  for (int k0 = 0; k0 < K; k0 += 32) {
    uint4 alo = *(const uint4*)(arow + k0 + akoff);
    uint4 ahi = *(const uint4*)(arow + k0 + 16 + akoff);
    v16bf a = make_v16(alo, ahi);
    v16bf b = make_v16(*(const uint4*)(brow + k0 + bkoff), *(const uint4*)(brow + k0 + 16 + bkoff));
    acc = __builtin_amdgcn_wmma_f32_16x16x32_bf16(false, a, false, b,
                                                  (short)0, acc, false, false);
  }
  float* crow = C + (size_t)(m0 + half * 8) * Hn + n0 + lid;
#pragma unroll
  for (int v = 0; v < 8; ++v) crow[(size_t)v * Hn] = acc[v];
}

// ---------------- edge scatter: agg[dst] += t[src] * dinv[src]*dinv[dst] ----------------
__global__ void edge_agg_k(const float* __restrict__ t, float* __restrict__ agg,
                           const int* __restrict__ src, const int* __restrict__ dst,
                           const float* __restrict__ dinv) {
  int e = blockIdx.x;
  int f = threadIdx.x;
  int s = src[e], d = dst[e];
  float nrm = dinv[s] * dinv[d];
  atomicAdd(&agg[(size_t)d * Hn + f], t[(size_t)s * Hn + f] * nrm);
}

// self-loop + bias + relu + bf16 convert (layer 1 epilogue)
__global__ void post_relu_bf16_k(const float* __restrict__ agg, const float* __restrict__ t,
                                 const float* __restrict__ dinv, const float* __restrict__ b,
                                 bf16* __restrict__ hb) {
  int i = blockIdx.x, f = threadIdx.x;
  float di = dinv[i];
  float v = agg[(size_t)i * Hn + f] + t[(size_t)i * Hn + f] * di * di + b[f];
  hb[(size_t)i * Hn + f] = (bf16)fmaxf(v, 0.0f);
}

// self-loop + bias + segment_max via encoded atomicMax (layer 2 epilogue + pool)
__global__ void pool_max_k(const float* __restrict__ agg, const float* __restrict__ t,
                           const float* __restrict__ dinv, const float* __restrict__ b,
                           const int* __restrict__ batch, unsigned* __restrict__ penc) {
  int i = blockIdx.x, f = threadIdx.x;
  float di = dinv[i];
  float v = agg[(size_t)i * Hn + f] + t[(size_t)i * Hn + f] * di * di + b[f];
  atomicMax(&penc[(size_t)batch[i] * Hn + f], enc_f32(v));
}

// row-wise log_softmax over [G, 256]; one block per row
__global__ void log_softmax_k(const unsigned* __restrict__ penc, float* __restrict__ out) {
  __shared__ float red[Hn];
  int g = blockIdx.x, f = threadIdx.x;
  float v = dec_f32(penc[(size_t)g * Hn + f]);
  red[f] = v;
  __syncthreads();
  for (int s = Hn / 2; s > 0; s >>= 1) {
    if (f < s) red[f] = fmaxf(red[f], red[f + s]);
    __syncthreads();
  }
  float m = red[0];
  __syncthreads();
  float e = expf(v - m);
  red[f] = e;
  __syncthreads();
  for (int s = Hn / 2; s > 0; s >>= 1) {
    if (f < s) red[f] += red[f + s];
    __syncthreads();
  }
  float lse = logf(red[0]);
  out[(size_t)g * Hn + f] = v - m - lse;
}

// ---------------- launch ----------------
extern "C" void kernel_launch(void* const* d_in, const int* in_sizes, int n_in,
                              void* d_out, int out_size, void* d_ws, size_t ws_size,
                              hipStream_t stream) {
  const float* x  = (const float*)d_in[0];   // [N, 128]
  const float* W1 = (const float*)d_in[1];   // [128, 256]
  const float* b1 = (const float*)d_in[2];   // [256]
  const float* W2 = (const float*)d_in[3];   // [256, 256]
  const float* b2 = (const float*)d_in[4];   // [256]
  const int*   ei = (const int*)d_in[5];     // [2, E]
  const int*   batch = (const int*)d_in[6];  // [N]
  const int* src = ei;
  const int* dst = ei + En;
  float* out = (float*)d_out;                // [512, 256]

  // workspace carve-out (~142 MB total)
  char* base = (char*)d_ws;
  size_t off = 0;
  auto carve = [&](size_t bytes) -> char* {
    char* p = base + off;
    off = (off + bytes + 255) & ~(size_t)255;
    return p;
  };
  bf16*     xb   = (bf16*)carve((size_t)Nn * FIN * sizeof(bf16));
  bf16*     w1t  = (bf16*)carve((size_t)Hn * FIN * sizeof(bf16));
  bf16*     w2t  = (bf16*)carve((size_t)Hn * Hn * sizeof(bf16));
  bf16*     hb   = (bf16*)carve((size_t)Nn * Hn * sizeof(bf16));
  float*    t    = (float*)carve((size_t)Nn * Hn * sizeof(float));
  float*    agg  = (float*)carve((size_t)Nn * Hn * sizeof(float));
  float*    dinv = (float*)carve((size_t)Nn * sizeof(float));
  unsigned* penc = (unsigned*)carve((size_t)Gn * Hn * sizeof(unsigned));

  const int B = 256;

  // degree -> dinv (deg includes self-loop: init 1)
  fill_f32_k<<<256, B, 0, stream>>>(dinv, 1.0f, (long)Nn);
  deg_edges_k<<<(En + B - 1) / B, B, 0, stream>>>(dinv, dst, En);
  rsqrt_inplace_k<<<(Nn + B - 1) / B, B, 0, stream>>>(dinv, Nn);

  // bf16 conversions
  f32_to_bf16_k<<<2048, B, 0, stream>>>(x, xb, (long)Nn * FIN);
  transpose_bf16_k<<<(FIN * Hn + B - 1) / B, B, 0, stream>>>(W1, w1t, FIN, Hn);
  transpose_bf16_k<<<(Hn * Hn + B - 1) / B, B, 0, stream>>>(W2, w2t, Hn, Hn);

  // layer 1: t = x @ W1 (WMMA), aggregate, epilogue -> hb
  gemm_bf16_wmma_k<<<dim3(Nn / 16, 2), B, 0, stream>>>(xb, w1t, t, FIN);
  fill_f32_k<<<2048, B, 0, stream>>>(agg, 0.0f, (long)Nn * Hn);
  edge_agg_k<<<En, B, 0, stream>>>(t, agg, src, dst, dinv);
  post_relu_bf16_k<<<Nn, B, 0, stream>>>(agg, t, dinv, b1, hb);

  // layer 2: t = hb @ W2 (WMMA), aggregate
  gemm_bf16_wmma_k<<<dim3(Nn / 16, 2), B, 0, stream>>>(hb, w2t, t, Hn);
  fill_f32_k<<<2048, B, 0, stream>>>(agg, 0.0f, (long)Nn * Hn);
  edge_agg_k<<<En, B, 0, stream>>>(t, agg, src, dst, dinv);

  // pool (segment max) + log_softmax
  fill_u32_k<<<64, B, 0, stream>>>(penc, ENC_NEG_INF, (long)Gn * Hn);
  pool_max_k<<<Nn, B, 0, stream>>>(agg, t, dinv, b2, batch, penc);
  log_softmax_k<<<Gn, B, 0, stream>>>(penc, out);
}
